// VmambaBlock_67688684585351
// MI455X (gfx1250) — matrix operand order repaired and checked by reading, not compiled, not run on hardware
//
#include <hip/hip_runtime.h>
#include <stdint.h>

// Fixed problem dims from setup_inputs()
#define Bn 4
#define Hh 64
#define Ww 64
#define Cc 96
#define Dd 192
#define Nn 16
#define Rr 6
#define Kk 4
#define Ll (Hh*Ww)        // 4096
#define Mrows (Bn*Ll)     // 16384
#define BKL (Bn*Kk*Ll)    // 65536
#define CHUNK 128
#define NCH (Ll/CHUNK)    // 32
#define SEQS (Bn*Kk*Dd*Nn) // 49152 independent recurrences

typedef __attribute__((ext_vector_type(16))) __bf16 v16bf;
typedef __attribute__((ext_vector_type(8)))  float  v8f;
typedef __attribute__((ext_vector_type(4)))  unsigned int u32x4;

union BFrag { v16bf v; u32x4 q[2]; };

__device__ __forceinline__ unsigned short f2bf(float f) {
  union { float f; uint32_t u; } v; v.f = f;
  uint32_t r = v.u + 0x7FFFu + ((v.u >> 16) & 1u);   // round-to-nearest-even
  return (unsigned short)(r >> 16);
}

__device__ __forceinline__ float wred(float s) {     // full wave32 reduce
  #pragma unroll
  for (int m = 16; m >= 1; m >>= 1) s += __shfl_xor(s, m, 32);
  return s;
}

// scan position l (direction k) -> spatial index into [H*W]; also the output position
__device__ __forceinline__ int map_l(int k, int l) {
  if (k == 0) return l;
  if (k == 2) return (Ll - 1) - l;
  int t = (k == 1) ? l : (Ll - 1) - l;   // x_wh ordering: l = w*H + h
  int w = t >> 6;
  int h = t & 63;
  return h * Ww + w;
}

// ---------------- fused double LayerNorm: xn (f32) + xv (bf16) ----------------
__global__ void ln2_kernel(const float* __restrict__ x,
                           const float* __restrict__ g1, const float* __restrict__ b1,
                           const float* __restrict__ g2, const float* __restrict__ b2,
                           float* __restrict__ xn, unsigned short* __restrict__ xv)
{
  int wid = blockIdx.x * 8 + (threadIdx.x >> 5);
  int lane = threadIdx.x & 31;
  const float* row = x + (size_t)wid * Cc;
  float v[3];
  #pragma unroll
  for (int j = 0; j < 3; j++) v[j] = row[lane + 32*j];
  float mean = wred(v[0]+v[1]+v[2]) * (1.0f/Cc);
  float q = 0.f;
  #pragma unroll
  for (int j = 0; j < 3; j++) { float d = v[j]-mean; q += d*d; }
  float inv = rsqrtf(wred(q)*(1.0f/Cc) + 1e-5f);
  float n[3];
  #pragma unroll
  for (int j = 0; j < 3; j++) n[j] = (v[j]-mean)*inv*g1[lane+32*j] + b1[lane+32*j];
  float mean2 = wred(n[0]+n[1]+n[2]) * (1.0f/Cc);
  float q2 = 0.f;
  #pragma unroll
  for (int j = 0; j < 3; j++) { float d = n[j]-mean2; q2 += d*d; }
  float inv2 = rsqrtf(wred(q2)*(1.0f/Cc) + 1e-5f);
  float* xnr = xn + (size_t)wid * Cc;
  unsigned short* xvr = xv + (size_t)wid * Cc;
  #pragma unroll
  for (int j = 0; j < 3; j++) {
    xnr[lane+32*j] = n[j];
    float o = (n[j]-mean2)*inv2*g2[lane+32*j] + b2[lane+32*j];
    xvr[lane+32*j] = f2bf(o);
  }
}

__global__ void f32_to_bf16_kernel(const float* __restrict__ s,
                                   unsigned short* __restrict__ d, int n) {
  int i = blockIdx.x*256 + threadIdx.x;
  if (i < n) d[i] = f2bf(s[i]);
}

__global__ void zero_kernel(float* __restrict__ p, int n) {
  int i = blockIdx.x*256 + threadIdx.x;
  if (i < n) p[i] = 0.0f;
}

// -------- bf16 WMMA GEMM: C[M,N] = A[M,K] @ Bw[N,K]^T (+ add0 + add1) --------
// One wave per 16x(16*NT) strip; A fragment reused across NT tiles. Fragments
// are loaded in-loop (no manual double buffering -- avoids VGPR spills; 8
// waves/SIMD occupancy hides the load latency instead).
template<int NT>
__global__ void __launch_bounds__(256)
gemm_bf16_wmma(const unsigned short* __restrict__ A,
               const unsigned short* __restrict__ Bw,
               float* __restrict__ Cout,
               const float* __restrict__ add0,
               const float* __restrict__ add1,
               int M, int Nc, int Kd)
{
  int wid  = blockIdx.x * 8 + (threadIdx.x >> 5);
  int lane = threadIdx.x & 31;
  int nGroups = Nc / (16*NT);
  int mT = wid / nGroups, nG = wid % nGroups;
  if (mT * 16 >= M) return;

  v8f acc[NT];
  #pragma unroll
  for (int t = 0; t < NT; t++)
    #pragma unroll
    for (int r = 0; r < 8; r++) acc[t][r] = 0.0f;

  int laneHi = lane >> 4;
  int laneLo = lane & 15;
  for (int kk = 0; kk < Kd; kk += 32) {
    // A 16x32: lane laneLo = row M; K lo/hi halves split on lane bit4
    BFrag af;
    const unsigned short* ap = A + (size_t)(mT*16 + laneLo)*Kd + kk + (laneHi << 3);
    af.q[0] = *(const u32x4*)ap;         // K +0..7 / +8..15
    af.q[1] = *(const u32x4*)(ap + 16);  // K +16..23 / +24..31
    #pragma unroll
    for (int t = 0; t < NT; t++) {
      // B 32x16: lane laneLo = column N; 16 K-values per lane, halves on bit4
      BFrag bf;
      const unsigned short* bp = Bw + (size_t)((nG*NT + t)*16 + laneLo)*Kd + kk + (laneHi << 3);
      bf.q[0] = *(const u32x4*)bp;
      bf.q[1] = *(const u32x4*)(bp + 16);
      acc[t] = __builtin_amdgcn_wmma_f32_16x16x32_bf16(
          false, af.v, false, bf.v, (short)0, acc[t], false, false);
    }
  }
  #pragma unroll
  for (int t = 0; t < NT; t++) {
    int nBase = (nG*NT + t)*16;
    #pragma unroll
    for (int r = 0; r < 8; r++) {
      int m = mT*16 + r + (laneHi << 3);
      int n = nBase + laneLo;
      size_t idx = (size_t)m * Nc + n;
      float vv = acc[t][r];
      if (add0) vv += add0[idx] + add1[idx];
      Cout[idx] = vv;
    }
  }
}

// ---------------- depthwise 3x3 conv + bias + SiLU; [B,L,2D] -> [B,D,L] ----------------
__global__ void conv_silu_kernel(const float* __restrict__ xz,
                                 const float* __restrict__ cw,
                                 const float* __restrict__ cb,
                                 float* __restrict__ xc)
{
  int bhw = blockIdx.x;
  int d = threadIdx.x;
  int b = bhw / Ll, hw = bhw % Ll;
  int h = hw >> 6, w = hw & 63;
  float acc = cb[d];
  #pragma unroll
  for (int dy = -1; dy <= 1; dy++) {
    int hh = h + dy; if (hh < 0 || hh >= Hh) continue;
    #pragma unroll
    for (int dx = -1; dx <= 1; dx++) {
      int ww2 = w + dx; if (ww2 < 0 || ww2 >= Ww) continue;
      acc += xz[((size_t)(b*Ll + hh*Ww + ww2))*(2*Dd) + d] * cw[d*9 + (dy+1)*3 + (dx+1)];
    }
  }
  acc = acc / (1.0f + __expf(-acc));       // SiLU
  xc[((size_t)(b*Dd + d))*Ll + hw] = acc;
}

// -------- x_proj (38 outs) + dt_proj + softplus, one wave per (b,k,l) --------
__global__ void xproj_kernel(const float* __restrict__ xc,    // [B,D,L]
                             const float* __restrict__ xw,    // [K,38,D]
                             const float* __restrict__ dtw,   // [K,D,R]
                             const float* __restrict__ dtb,   // [K,D]
                             float* __restrict__ dtbuf,       // [BKL,D]
                             float* __restrict__ Bbuf,        // [BKL,16]
                             float* __restrict__ Cbuf)        // [BKL,16]
{
  int g = blockIdx.x*8 + (threadIdx.x >> 5);
  int lane = threadIdx.x & 31;
  int l  = g & (Ll-1);
  int bk = g / Ll;
  int k = bk & 3, b = bk >> 2;
  int l0 = map_l(k, l);
  float u[6];
  #pragma unroll
  for (int j = 0; j < 6; j++) u[j] = xc[((size_t)(b*Dd) + lane + 32*j)*Ll + l0];
  float dtr[6];
  #pragma unroll
  for (int c = 0; c < 38; c++) {
    float p = 0.f;
    #pragma unroll
    for (int j = 0; j < 6; j++) p += u[j] * xw[((size_t)(k*38 + c))*Dd + lane + 32*j];
    p = wred(p);
    if (c < 6)        dtr[c] = p;
    else if (c < 22)  { if (lane == 0) Bbuf[(size_t)g*16 + (c-6)]  = p; }
    else              { if (lane == 0) Cbuf[(size_t)g*16 + (c-22)] = p; }
  }
  #pragma unroll
  for (int j = 0; j < 6; j++) {
    int d = lane + 32*j;
    float a = dtb[k*Dd + d];
    #pragma unroll
    for (int r = 0; r < 6; r++) a += dtr[r] * dtw[((size_t)(k*Dd) + d)*Rr + r];
    dtbuf[(size_t)g*Dd + d] = (a > 20.0f) ? a : log1pf(__expf(a));  // softplus
  }
}

// ======== chunked selective scan: h_l = a_l*h_{l-1} + b_l (3 phases) ========
// Phase 1: per (b,k,d,n,chunk) compute chunk decay product and partial state.
// Wave org: lanes 0-15 -> states of d0, lanes 16-31 -> states of d1.
__global__ void scan_phase1_kernel(const float* __restrict__ xc,     // [B,D,L]
                                   const float* __restrict__ dtbuf,  // [BK,L,D]
                                   const float* __restrict__ Bbuf,   // [BK,L,16]
                                   const float* __restrict__ A_logs, // [K,D,N]
                                   float* __restrict__ Aprod,        // [BK,D,N,NCH]
                                   float* __restrict__ Hpart)        // [BK,D,N,NCH]
{
  int wid  = blockIdx.x*8 + (threadIdx.x >> 5);
  int lane = threadIdx.x & 31;
  int half = lane >> 4, n = lane & 15;
  int c  = wid & (NCH-1);
  int t  = wid / NCH;             // (bk, dpair)
  int dp = t % (Dd/2);
  int bk = t / (Dd/2);
  int d  = dp*2 + half;
  int k  = bk & 3, b = bk >> 2;
  float An = -__expf(A_logs[((size_t)(k*Dd) + d)*Nn + n]);
  const float* xcd = xc + ((size_t)(b*Dd) + d)*Ll;
  float h = 0.0f, ap = 1.0f;
  int lbase = c * CHUNK;
  for (int j = 0; j < CHUNK; j++) {
    int l = lbase + j;
    float dt = dtbuf[((size_t)bk*Ll + l)*Dd + d];
    float u  = xcd[map_l(k, l)];
    float Bt = Bbuf[((size_t)bk*Ll + l)*16 + n];
    float a  = __expf(dt*An);
    h  = a*h + (dt*u)*Bt;
    ap *= a;
  }
  size_t idx = (((size_t)bk*Dd + d)*Nn + n)*NCH + c;
  Aprod[idx] = ap;
  Hpart[idx] = h;
}

// Phase 2: exact initial state per chunk via tiny sequential combine (depth 32).
__global__ void scan_phase2_kernel(const float* __restrict__ Aprod,
                                   const float* __restrict__ Hpart,
                                   float* __restrict__ Hinit)
{
  int i = blockIdx.x*256 + threadIdx.x;   // over SEQS = BK*D*N
  if (i >= SEQS) return;
  const float* ap = Aprod + (size_t)i*NCH;
  const float* hp = Hpart + (size_t)i*NCH;
  float*       hi = Hinit + (size_t)i*NCH;
  float h = 0.0f;
  #pragma unroll
  for (int c = 0; c < NCH; c++) {
    hi[c] = h;                 // state entering chunk c
    h = ap[c]*h + hp[c];
  }
}

// Phase 3: replay each chunk from its exact initial state, emit y into y[b,l0,d].
__global__ void scan_phase3_kernel(const float* __restrict__ xc,
                                   const float* __restrict__ dtbuf,
                                   const float* __restrict__ Bbuf,
                                   const float* __restrict__ Cbuf,
                                   const float* __restrict__ A_logs,
                                   const float* __restrict__ Ds,      // [K,D]
                                   const float* __restrict__ Hinit,
                                   float* __restrict__ yc)            // [B,L,D] pre-zeroed
{
  int wid  = blockIdx.x*8 + (threadIdx.x >> 5);
  int lane = threadIdx.x & 31;
  int half = lane >> 4, n = lane & 15;
  int c  = wid & (NCH-1);
  int t  = wid / NCH;
  int dp = t % (Dd/2);
  int bk = t / (Dd/2);
  int d  = dp*2 + half;
  int k  = bk & 3, b = bk >> 2;
  float An = -__expf(A_logs[((size_t)(k*Dd) + d)*Nn + n]);
  float Dv = Ds[k*Dd + d];
  const float* xcd = xc + ((size_t)(b*Dd) + d)*Ll;
  float h = Hinit[(((size_t)bk*Dd + d)*Nn + n)*NCH + c];
  int lbase = c * CHUNK;
  for (int j = 0; j < CHUNK; j++) {
    int l = lbase + j;
    float dt = dtbuf[((size_t)bk*Ll + l)*Dd + d];
    int l0 = map_l(k, l);
    float u  = xcd[l0];
    float Bt = Bbuf[((size_t)bk*Ll + l)*16 + n];
    float Ct = Cbuf[((size_t)bk*Ll + l)*16 + n];
    h = __expf(dt*An)*h + (dt*u)*Bt;
    float y = h*Ct;
    y += __shfl_xor(y, 1, 32); y += __shfl_xor(y, 2, 32);   // reduce within 16-lane half
    y += __shfl_xor(y, 4, 32); y += __shfl_xor(y, 8, 32);
    if (n == 0) atomicAdd(&yc[((size_t)b*Ll + l0)*Dd + d], y + u*Dv);
  }
}

// ------- out_norm + silu(z) gate + bf16 pack, one wave per (b,l) -------
__global__ void outnorm_gate_kernel(const float* __restrict__ yc,
                                    const float* __restrict__ g, const float* __restrict__ bb,
                                    const float* __restrict__ xz,
                                    unsigned short* __restrict__ yb)
{
  int wid = blockIdx.x*8 + (threadIdx.x >> 5);
  int lane = threadIdx.x & 31;
  const float* row = yc + (size_t)wid * Dd;
  float v[6];
  #pragma unroll
  for (int j = 0; j < 6; j++) v[j] = row[lane + 32*j];
  float mean = wred(v[0]+v[1]+v[2]+v[3]+v[4]+v[5]) * (1.0f/Dd);
  float q = 0.f;
  #pragma unroll
  for (int j = 0; j < 6; j++) { float dd = v[j]-mean; q += dd*dd; }
  float inv = rsqrtf(wred(q)*(1.0f/Dd) + 1e-5f);
  unsigned short* out = yb + (size_t)wid * Dd;
  #pragma unroll
  for (int j = 0; j < 6; j++) {
    int d = lane + 32*j;
    float y = (v[j]-mean)*inv*g[d] + bb[d];
    float z = xz[(size_t)wid*(2*Dd) + Dd + d];
    y *= z / (1.0f + __expf(-z));
    out[d] = f2bf(y);
  }
}

extern "C" void kernel_launch(void* const* d_in, const int* in_sizes, int n_in,
                              void* d_out, int out_size, void* d_ws, size_t ws_size,
                              hipStream_t stream)
{
  (void)in_sizes; (void)n_in; (void)out_size; (void)ws_size;
  const float* x      = (const float*)d_in[0];
  const float* n1g    = (const float*)d_in[1];
  const float* n1b    = (const float*)d_in[2];
  const float* lng    = (const float*)d_in[3];
  const float* lnb    = (const float*)d_in[4];
  const float* winp   = (const float*)d_in[5];   // [384,96]
  const float* convw  = (const float*)d_in[6];   // [192,1,3,3]
  const float* convb  = (const float*)d_in[7];
  const float* xprojw = (const float*)d_in[8];   // [4,38,192]
  const float* dtw    = (const float*)d_in[9];   // [4,192,6]
  const float* dtbv   = (const float*)d_in[10];  // [4,192]
  const float* Alogs  = (const float*)d_in[11];  // [4,192,16]
  const float* Dsv    = (const float*)d_in[12];  // [4,192]
  const float* ong    = (const float*)d_in[13];
  const float* onb    = (const float*)d_in[14];
  const float* woutp  = (const float*)d_in[15];  // [96,192]

  char* p = (char*)d_ws;
  auto take = [&](size_t bytes) -> void* {
    void* r = (void*)p; p += (bytes + 255) & ~(size_t)255; return r;
  };
  float*          xn  = (float*)take((size_t)Mrows*Cc*4);
  unsigned short* xv  = (unsigned short*)take((size_t)Mrows*Cc*2);
  unsigned short* wib = (unsigned short*)take((size_t)2*Dd*Cc*2);
  unsigned short* wob = (unsigned short*)take((size_t)Cc*Dd*2);
  float*          xz  = (float*)take((size_t)Mrows*2*Dd*4);
  float*          xc  = (float*)take((size_t)Bn*Dd*Ll*4);
  float*          dtb = (float*)take((size_t)BKL*Dd*4);
  float*          Bb  = (float*)take((size_t)BKL*16*4);
  float*          Cb  = (float*)take((size_t)BKL*16*4);
  float*          yc  = (float*)take((size_t)Mrows*Dd*4);
  unsigned short* yb  = (unsigned short*)take((size_t)Mrows*Dd*2);
  float*          Ap  = (float*)take((size_t)SEQS*NCH*4);
  float*          Hp  = (float*)take((size_t)SEQS*NCH*4);
  float*          Hi  = (float*)take((size_t)SEQS*NCH*4);

  ln2_kernel<<<Mrows/8, 256, 0, stream>>>(x, n1g, n1b, lng, lnb, xn, xv);
  f32_to_bf16_kernel<<<(2*Dd*Cc + 255)/256, 256, 0, stream>>>(winp, wib, 2*Dd*Cc);
  f32_to_bf16_kernel<<<(Cc*Dd + 255)/256, 256, 0, stream>>>(woutp, wob, Cc*Dd);

  // in_proj: M=16384, N=384, K=96; NT=4 -> 6 n-groups, 6144 waves
  gemm_bf16_wmma<4><<<(1024*6)/8, 256, 0, stream>>>(xv, wib, xz, nullptr, nullptr,
                                                    Mrows, 2*Dd, Cc);
  conv_silu_kernel<<<Bn*Ll, Dd, 0, stream>>>(xz, convw, convb, xc);
  xproj_kernel<<<BKL/8, 256, 0, stream>>>(xc, xprojw, dtw, dtbv, dtb, Bb, Cb);
  zero_kernel<<<(Mrows*Dd + 255)/256, 256, 0, stream>>>(yc, Mrows*Dd);

  // chunked scan: 16*96*32 = 49152 waves in phases 1 and 3
  scan_phase1_kernel<<<(Bn*Kk*(Dd/2)*NCH)/8, 256, 0, stream>>>(xc, dtb, Bb, Alogs, Ap, Hp);
  scan_phase2_kernel<<<(SEQS + 255)/256, 256, 0, stream>>>(Ap, Hp, Hi);
  scan_phase3_kernel<<<(Bn*Kk*(Dd/2)*NCH)/8, 256, 0, stream>>>(xc, dtb, Bb, Cb, Alogs,
                                                               Dsv, Hi, yc);

  outnorm_gate_kernel<<<Mrows/8, 256, 0, stream>>>(yc, ong, onb, xz, yb);

  // out_proj fused with both residuals: d_out = x + xn + yb @ wob^T
  gemm_bf16_wmma<2><<<(1024*3)/8, 256, 0, stream>>>(yb, wob, (float*)d_out, x, xn,
                                                    Mrows, Cc, Dd);
}
